// SAXS_to_Contact_Map_10316511445632
// MI455X (gfx1250) — matrix operand order repaired and checked by reading, not compiled, not run on hardware
//
#include <hip/hip_runtime.h>
#include <hip/hip_bf16.h>

// Problem sizes
#define B_    1024
#define NIN   512
#define HID   64
#define NOUT  256
#define QKN   (NOUT * HID)   // 16384

typedef __attribute__((ext_vector_type(16))) _Float16 v16h;
typedef __attribute__((ext_vector_type(8)))  _Float16 v8h;
typedef __attribute__((ext_vector_type(8)))  float    v8f;
typedef __attribute__((ext_vector_type(4)))  float    v4f;
typedef __attribute__((ext_vector_type(4)))  unsigned u32x4;
typedef __attribute__((ext_vector_type(8)))  unsigned u32x8;

// ---------------------------------------------------------------------------
// CDNA5 async global->LDS copy (ASYNCcnt path), per CDNA5 ISA.
// LDS byte address = low 32 bits of the generic (flat) pointer, per the ISA
// aperture rule for LDS-aperture flat addresses.
// ---------------------------------------------------------------------------
static __device__ inline void async_load_b128(const void* gsrc, void* ldst) {
  asm volatile("global_load_async_to_lds_b128 %0, %1, off"
               :
               : "v"((unsigned)(unsigned long long)ldst),
                 "v"((unsigned long long)gsrc)
               : "memory");
}
static __device__ inline void wait_asynccnt0() {
  asm volatile("s_wait_asynccnt 0x0" ::: "memory");
}

// ---------------------------------------------------------------------------
// CDNA5 Tensor Data Mover: 2D f16 tile, global -> padded LDS (TENSORcnt path).
// D# per CDNA5 ISA (2-group form, <=2D tensor):
//   group0: [1:0]=count=1 | [63:32]=lds_addr | [120:64]=global_addr | [127:126]=type 2
//   group1: data_size=1 (2 bytes), pad_enable, pad_interval/pad_amount codes,
//           tensor_dim0/1, tile_dim0/1, tensor_dim0_stride (elements)
// Issue once per wave (TDM ignores EXEC); caller gates to one wave.
// ---------------------------------------------------------------------------
static __device__ inline void tdm_load_2d_f16(const void* gsrc, void* ldst,
                                              unsigned tile_d0, unsigned tile_d1,
                                              unsigned stride_e,
                                              unsigned pad_interval_code,
                                              unsigned pad_amount_code) {
  const unsigned long long ga = (unsigned long long)gsrc;
  u32x4 g0;
  g0[0] = 1u;                                                   // count=1
  g0[1] = (unsigned)(unsigned long long)ldst;                   // LDS byte address
  g0[2] = (unsigned)ga;                                         // global addr [31:0]
  g0[3] = (unsigned)((ga >> 32) & 0x01FFFFFFull) | 0x80000000u; // addr[56:32] | type=2
  u32x8 g1;
  g1[0] = (1u << 16) | (1u << 20) |                             // data_size=2B, pad_enable
          (pad_interval_code << 22) | (pad_amount_code << 25);
  g1[1] = (stride_e & 0xFFFFu) << 16;                           // tensor_dim0[15:0]
  g1[2] = (stride_e >> 16) | ((tile_d1 & 0xFFFFu) << 16);       // tensor_dim0[31:16], dim1[15:0]
  g1[3] = (tile_d1 >> 16) | ((tile_d0 & 0xFFFFu) << 16);        // tensor_dim1[31:16], tile_dim0
  g1[4] = tile_d1 & 0xFFFFu;                                    // tile_dim1 (tile_dim2 = 0)
  g1[5] = stride_e;                                             // tensor_dim0_stride[31:0]
  g1[6] = 0u;                                                   // stride hi / dim1_stride lo
  g1[7] = 0u;                                                   // dim1_stride hi
  asm volatile("tensor_load_to_lds %0, %1" :: "s"(g0), "s"(g1) : "memory");
}

// ---------------------------------------------------------------------------
// WMMA fragment loaders per CDNA5 ISA 7.12.2 (16-bit, 16x16x32)
// A (16x32, MxK): lanes 0-15 -> M=lane, halves 0-7 = K 0-7, halves 8-15 = K 16-23
//                 lanes 16-31 -> M=lane-16, halves 0-7 = K 8-15, halves 8-15 = K 24-31
// B (32x16, KxN): N = lane&15, K order same as A (halves 0-7 = K 8h..8h+7, halves 8-15 = K 16+8h..)
// C/D (16x16 f32): N = lane&15, M = 8*(lane>=16) + vgpr_index
// ---------------------------------------------------------------------------
static __device__ inline v16h frag_a(const _Float16* base, int stride, int m0, int k0, int lane) {
  int m  = m0 + (lane & 15);
  int kb = k0 + ((lane >> 4) << 3);          // +0 for lanes 0-15, +8 for lanes 16-31
  const _Float16* p = base + m * stride + kb;
  v8h lo = *(const v8h*)(p);                 // K = kb .. kb+7
  v8h hi = *(const v8h*)(p + 16);            // K = kb+16 .. kb+23
  v16h a;
#pragma unroll
  for (int i = 0; i < 8; ++i) { a[i] = lo[i]; a[i + 8] = hi[i]; }
  return a;
}

static __device__ inline v16h frag_b(const _Float16* base, int stride, int k0, int n0, int lane) {
  const _Float16* p = base + (k0 + ((lane >> 4) << 3)) * stride + n0 + (lane & 15);
  v16h b;
#pragma unroll
  for (int i = 0; i < 8; ++i) { b[i] = p[i * stride]; b[i + 8] = p[(i + 16) * stride]; }
  return b;
}

// ---------------------------------------------------------------------------
// Kernel 1: LayerNorm, f32 in -> f16 out. One block per batch row (512 elems).
// ---------------------------------------------------------------------------
__global__ __launch_bounds__(256) void ln_f16_kernel(const float* __restrict__ x,
                                                     const float* __restrict__ gamma,
                                                     const float* __restrict__ beta,
                                                     _Float16* __restrict__ h) {
  __shared__ float red[256];
  const int b = blockIdx.x, t = threadIdx.x;
  const float a0 = x[(size_t)b * NIN + t];
  const float a1 = x[(size_t)b * NIN + 256 + t];

  red[t] = a0 + a1;
  __syncthreads();
  for (int off = 128; off > 0; off >>= 1) {
    if (t < off) red[t] += red[t + off];
    __syncthreads();
  }
  const float mu = red[0] * (1.0f / NIN);
  __syncthreads();

  const float d0 = a0 - mu, d1 = a1 - mu;
  red[t] = d0 * d0 + d1 * d1;
  __syncthreads();
  for (int off = 128; off > 0; off >>= 1) {
    if (t < off) red[t] += red[t + off];
    __syncthreads();
  }
  const float rstd = rsqrtf(red[0] * (1.0f / NIN) + 1e-5f);

  h[(size_t)b * NIN + t]       = (_Float16)(d0 * rstd * gamma[t]       + beta[t]);
  h[(size_t)b * NIN + 256 + t] = (_Float16)(d1 * rstd * gamma[t + 256] + beta[t + 256]);
}

// ---------------------------------------------------------------------------
// Kernel 2: QKV projection GEMM. C(1024x16384) = h(1024x512,f16) @ W(512x16384,f32)+bias.
// One block per 64-wide N strip; M looped inside; the W strip is re-read for each 128-row M block.
// h tile (128x32 f16, row stride 512 -> LDS stride 40, pad 4 DW / 16 DW) staged
// by the Tensor Data Mover (one descriptor per k-step, issued by wave 0);
// W tile staged through VGPRs with f32->f16 conversion.
// ---------------------------------------------------------------------------
__global__ __launch_bounds__(256) void qkv_gemm_kernel(const _Float16* __restrict__ h,
                                                       const float* __restrict__ W,
                                                       const float* __restrict__ bias,
                                                       _Float16* __restrict__ outb) {
  __shared__ _Float16 sA[128 * 40];   // 128 x 32 tile of h, stride 40 halves
  __shared__ _Float16 sB[32 * 72];    // 32 x 64 tile of W (f16), stride 72 halves
  const int n0   = blockIdx.x * 64;
  const int t    = threadIdx.x;
  const int lane = t & 31;
  const int wave = t >> 5;

  for (int m0 = 0; m0 < B_; m0 += 128) {
    v8f acc[4] = {};
    for (int k0 = 0; k0 < NIN; k0 += 32) {
      __syncthreads();
      if (wave == 0) {  // TDM: whole 128x32 h tile in one descriptor
        // rows of 32 halves = 16 DWORDs (pad_interval code 3), pad 4 DWORDs (code 3)
        tdm_load_2d_f16(h + (size_t)m0 * NIN + k0, sA,
                        /*tile_d0=*/32, /*tile_d1=*/128, /*stride=*/NIN,
                        /*pad_interval=*/3, /*pad_amount=*/3);
      }
      {  // stage W tile with f32->f16 conversion: each thread moves 8 floats
        const int row = t >> 3, col = (t & 7) * 8;
        const v4f* src = (const v4f*)(W + (size_t)(k0 + row) * QKN + n0 + col);
        v4f w0 = src[0], w1 = src[1];
        v8h d;
#pragma unroll
        for (int i = 0; i < 4; ++i) { d[i] = (_Float16)w0[i]; d[i + 4] = (_Float16)w1[i]; }
        *(v8h*)(sB + row * 72 + col) = d;
      }
      if (k0 + 32 < NIN)
        __builtin_prefetch(W + (size_t)(k0 + 32 + (t >> 3)) * QKN + n0 + (t & 7) * 8, 0, 0);
      if (wave == 0) __builtin_amdgcn_s_wait_tensorcnt(0);
      __syncthreads();

      v16h a = frag_a(sA, 40, wave * 16, 0, lane);
#pragma unroll
      for (int nt = 0; nt < 4; ++nt) {
        v16h b = frag_b(sB, 72, 0, nt * 16, lane);
        acc[nt] = __builtin_amdgcn_wmma_f32_16x16x32_f16(
            false, a, false, b, (short)0, acc[nt], false, false);
      }
    }
    // epilogue: add bias, convert to f16, store (B, NOUT*HID) layout
#pragma unroll
    for (int nt = 0; nt < 4; ++nt) {
      const int gn = n0 + nt * 16 + (lane & 15);
      const float bv = bias[gn];
#pragma unroll
      for (int r = 0; r < 8; ++r) {
        const int m = wave * 16 + ((lane >> 4) << 3) + r;
        outb[(size_t)(m0 + m) * QKN + gn] = (_Float16)(acc[nt][r] + bv);
      }
    }
  }
}

// ---------------------------------------------------------------------------
// Kernel 3: fused attention + output projection, one block per batch element.
// LDS plan (dynamic, ~244 KB -- legal on CDNA5's 320 KB WGP LDS):
//   [0      ) Qs  256x72 f16   (later aliased by As 256x72)
//   [36864  ) Kt   64x264 f16  (K transposed; later aliased by Wos 64x264)
//   [70656  ) Vs  256x72 f16
//   [107520 ) Ps  256x264 f16  (S logits, then softmax probs)
//   [242688 ) bos 256 f32
// Q and V staged with async global->LDS; K transposed through VGPRs.
// ---------------------------------------------------------------------------
#define ATTN_LDS_BYTES 243712

__global__ __launch_bounds__(256) void attn_kernel(const _Float16* __restrict__ Qg,
                                                   const _Float16* __restrict__ Kg,
                                                   const _Float16* __restrict__ Vg,
                                                   const float* __restrict__ Wo,
                                                   const float* __restrict__ bo,
                                                   float* __restrict__ out) {
  extern __shared__ char smem[];
  _Float16* Qs  = (_Float16*)(smem);
  _Float16* Kt  = (_Float16*)(smem + 36864);
  _Float16* Vs  = (_Float16*)(smem + 70656);
  _Float16* Ps  = (_Float16*)(smem + 107520);
  _Float16* As  = (_Float16*)(smem);            // alias of Qs (dead after S phase)
  _Float16* Wos = (_Float16*)(smem + 36864);    // alias of Kt (dead after S phase)
  float*    bos = (float*)(smem + 242688);

  const int b = blockIdx.x, t = threadIdx.x, lane = t & 31, wave = t >> 5;

  // ---- stage Q, K^T, V into LDS (thread t owns attention row t) ----
  {
    const _Float16* q = Qg + (size_t)b * QKN + t * HID;
    const _Float16* v = Vg + (size_t)b * QKN + t * HID;
    _Float16* qd = Qs + t * 72;
    _Float16* vd = Vs + t * 72;
#pragma unroll
    for (int i = 0; i < 8; ++i) {
      async_load_b128(q + i * 8, qd + i * 8);
      async_load_b128(v + i * 8, vd + i * 8);
    }
    const v8h* k = (const v8h*)(Kg + (size_t)b * QKN + t * HID);
#pragma unroll
    for (int i = 0; i < 8; ++i) {
      v8h kk = k[i];
#pragma unroll
      for (int e = 0; e < 8; ++e) Kt[(i * 8 + e) * 264 + t] = kk[e];  // transpose
    }
    bos[t] = bo[t];
  }
  wait_asynccnt0();
  __syncthreads();

  // ---- S = (Q @ K^T) * HID^-0.5 -> Ps ----  (256x256, 16x16 tiles, K=64)
  for (int tb = 0; tb < 2; ++tb) {
    const int it = wave * 2 + tb;
    for (int jt = 0; jt < 16; ++jt) {
      v8f acc = {};
#pragma unroll
      for (int ck = 0; ck < 2; ++ck) {
        v16h a  = frag_a(Qs, 72, it * 16, ck * 32, lane);
        v16h bf = frag_b(Kt, 264, ck * 32, jt * 16, lane);
        acc = __builtin_amdgcn_wmma_f32_16x16x32_f16(
            false, a, false, bf, (short)0, acc, false, false);
      }
      const int n = jt * 16 + (lane & 15);
#pragma unroll
      for (int r = 0; r < 8; ++r) {
        const int m = it * 16 + ((lane >> 4) << 3) + r;
        Ps[m * 264 + n] = (_Float16)(acc[r] * 0.125f);   // 1/sqrt(64)
      }
    }
  }
  __syncthreads();

  // ---- softmax over each row of Ps (thread t -> row t); stage Wo into freed Kt ----
  {
    _Float16* row = Ps + t * 264;
    float mx = -3.0e38f;
    for (int j = 0; j < NOUT; ++j) mx = fmaxf(mx, (float)row[j]);
    float s = 0.0f;
    for (int j = 0; j < NOUT; ++j) {
      const float e = __expf((float)row[j] - mx);
      s += e;
      row[j] = (_Float16)e;
    }
    const float inv = 1.0f / s;
    for (int j = 0; j < NOUT; ++j) row[j] = (_Float16)((float)row[j] * inv);
  }
#pragma unroll
  for (int i = 0; i < 64; ++i) {          // 16384 Wo elems / 256 threads
    const int idx = t * 64 + i;
    Wos[(idx >> 8) * 264 + (idx & 255)] = (_Float16)Wo[idx];
  }
  __syncthreads();

  // ---- A = P @ V -> As ----  (256x64, K=256)
  for (int tb = 0; tb < 2; ++tb) {
    const int it = wave * 2 + tb;
#pragma unroll
    for (int jt = 0; jt < 4; ++jt) {
      v8f acc = {};
#pragma unroll
      for (int ck = 0; ck < 8; ++ck) {
        v16h a  = frag_a(Ps, 264, it * 16, ck * 32, lane);
        v16h bf = frag_b(Vs, 72, ck * 32, jt * 16, lane);
        acc = __builtin_amdgcn_wmma_f32_16x16x32_f16(
            false, a, false, bf, (short)0, acc, false, false);
      }
      const int n = jt * 16 + (lane & 15);
#pragma unroll
      for (int r = 0; r < 8; ++r) {
        const int m = it * 16 + ((lane >> 4) << 3) + r;
        As[m * 72 + n] = (_Float16)acc[r];
      }
    }
  }
  __syncthreads();

  // ---- out = -relu(A @ Wo + bo) ----  (256x256, K=64)
  float* po = out + (size_t)b * (NOUT * NOUT);
  for (int tb = 0; tb < 2; ++tb) {
    const int it = wave * 2 + tb;
    for (int jt = 0; jt < 16; ++jt) {
      v8f acc = {};
#pragma unroll
      for (int ck = 0; ck < 2; ++ck) {
        v16h a  = frag_a(As, 72, it * 16, ck * 32, lane);
        v16h bf = frag_b(Wos, 264, ck * 32, jt * 16, lane);
        acc = __builtin_amdgcn_wmma_f32_16x16x32_f16(
            false, a, false, bf, (short)0, acc, false, false);
      }
      const int n = jt * 16 + (lane & 15);
      const float bb = bos[n];
#pragma unroll
      for (int r = 0; r < 8; ++r) {
        const int m = it * 16 + ((lane >> 4) << 3) + r;
        po[(size_t)m * NOUT + n] = -fmaxf(acc[r] + bb, 0.0f);
      }
    }
  }
}

// ---------------------------------------------------------------------------
// Launch: LN -> 3x QKV GEMM -> fused attention.
// Workspace: h f16 (1 MB) | Q f16 (32 MB) | K f16 (32 MB) | V f16 (32 MB)
// ---------------------------------------------------------------------------
extern "C" void kernel_launch(void* const* d_in, const int* in_sizes, int n_in,
                              void* d_out, int out_size, void* d_ws, size_t ws_size,
                              hipStream_t stream) {
  (void)in_sizes; (void)n_in; (void)out_size; (void)ws_size;
  const float* x  = (const float*)d_in[0];
  const float* lg = (const float*)d_in[1];
  const float* lb = (const float*)d_in[2];
  const float* Wq = (const float*)d_in[3];
  const float* bq = (const float*)d_in[4];
  const float* Wk = (const float*)d_in[5];
  const float* bk = (const float*)d_in[6];
  const float* Wv = (const float*)d_in[7];
  const float* bv = (const float*)d_in[8];
  const float* Wo = (const float*)d_in[9];
  const float* bo = (const float*)d_in[10];
  float* out = (float*)d_out;

  _Float16* hbuf = (_Float16*)d_ws;
  _Float16* qbuf = (_Float16*)((char*)d_ws + (size_t)(1u << 20));
  _Float16* kbuf = (_Float16*)((char*)d_ws + (size_t)(1u << 20) + (size_t)(32u << 20));
  _Float16* vbuf = (_Float16*)((char*)d_ws + (size_t)(1u << 20) + (size_t)(64u << 20));

  ln_f16_kernel<<<B_, 256, 0, stream>>>(x, lg, lb, hbuf);
  qkv_gemm_kernel<<<QKN / 64, 256, 0, stream>>>(hbuf, Wq, bq, qbuf);
  qkv_gemm_kernel<<<QKN / 64, 256, 0, stream>>>(hbuf, Wk, bk, kbuf);
  qkv_gemm_kernel<<<QKN / 64, 256, 0, stream>>>(hbuf, Wv, bv, vbuf);
  attn_kernel<<<B_, 256, ATTN_LDS_BYTES, stream>>>(qbuf, kbuf, vbuf, Wo, bo, out);
}
